// PairwiseRankingLoss_28329604284822
// MI455X (gfx1250) — compile-verified, matrix operand order checked by reading, not run on hardware
//
#include <hip/hip_runtime.h>
#include <hip/hip_bf16.h>

#define MARGIN 0.2f
#define N_ROWS 8192
#define DIM    1024

typedef __attribute__((ext_vector_type(16))) __bf16 v16bf;
typedef __attribute__((ext_vector_type(8)))  __bf16 v8bf;
typedef __attribute__((ext_vector_type(8)))  float  v8f;

// ---------------------------------------------------------------------------
// fp32 -> bf16 round-to-nearest-even (bit manipulation; finite data)
// ---------------------------------------------------------------------------
__device__ __forceinline__ unsigned short f2bf_rne(float x) {
    union { float f; unsigned int u; } v;
    v.f = x;
    unsigned int u = v.u;
    unsigned int r = u + 0x7FFFu + ((u >> 16) & 1u);
    return (unsigned short)(r >> 16);
}

// ---------------------------------------------------------------------------
// Kernel 1: diag[r] = <image_r, sentence_r>  (one wave32 per row, fp32)
// ---------------------------------------------------------------------------
__global__ __launch_bounds__(256) void diag_kernel(const float* __restrict__ img,
                                                   const float* __restrict__ sen,
                                                   float* __restrict__ diag) {
    const int wave = threadIdx.x >> 5;
    const int lane = threadIdx.x & 31;
    const int row  = blockIdx.x * 8 + wave;
    const float* a = img + (size_t)row * DIM;
    const float* b = sen + (size_t)row * DIM;
    float s = 0.0f;
    #pragma unroll 8
    for (int k = lane; k < DIM; k += 32) s += a[k] * b[k];
    #pragma unroll
    for (int off = 16; off > 0; off >>= 1) s += __shfl_down(s, off, 32);
    if (lane == 0) diag[row] = s;
}

// ---------------------------------------------------------------------------
// Kernel 1b: one-time fp32 -> bf16 matrix conversion (memory bound)
// ---------------------------------------------------------------------------
__global__ __launch_bounds__(256) void convert_bf16_kernel(const float* __restrict__ src,
                                                           unsigned short* __restrict__ dst) {
    const int base = (blockIdx.x * 256 + threadIdx.x) * 4;
    const float4 f = *(const float4*)(src + base);
    unsigned int p0 = (unsigned int)f2bf_rne(f.x) | ((unsigned int)f2bf_rne(f.y) << 16);
    unsigned int p1 = (unsigned int)f2bf_rne(f.z) | ((unsigned int)f2bf_rne(f.w) << 16);
    unsigned int* d = (unsigned int*)(dst + base);
    d[0] = p0;
    d[1] = p1;
}

// ---------------------------------------------------------------------------
// LDS tiles: rows x 32 K of bf16, row stride padded to 40 shorts (80 B)
// -> 16-row x 16-byte fragment loads are bank-conflict free.
// ---------------------------------------------------------------------------
#define TSTRIDE 40

// 16-bit A/B fragment per ISA layout: lane&15 -> row, regs -> K,
// lane[4] selects K halves {0..7,16..23} vs {8..15,24..31}
__device__ __forceinline__ v16bf load_frag(const unsigned short* tile,
                                           int rowInTile, int lane) {
    const int m  = lane & 15;
    const int kb = (lane >> 4) << 3; // 0 or 8 (shorts)
    const unsigned short* p = tile + (rowInTile + m) * TSTRIDE + kb;
    union { v16bf v; v8bf h[2]; } u;
    u.h[0] = *(const v8bf*)p;        // K = kb .. kb+7
    u.h[1] = *(const v8bf*)(p + 16); // K = kb+16 .. kb+23
    return u.v;
}

// ---------------------------------------------------------------------------
// Kernel 2 (fast path): bf16 panels staged with async-to-LDS copies.
//   block = 256 threads = 8 waves, block tile = 256x128,
//   wave subtile = 64x64 (waves arranged 4 row-groups x 2 col-groups)
// ---------------------------------------------------------------------------
template <int CHUNKS>
__device__ __forceinline__ void async_tile_load(const unsigned short* __restrict__ srcBase,
                                                int srcRowBase, int ks,
                                                const unsigned short* dstLds, int tid) {
    const unsigned ldsBase = (unsigned)(unsigned long long)dstLds;
    #pragma unroll
    for (int q = 0; q < CHUNKS; ++q) {
        const int chunk = tid + q * 256;      // 16-byte chunks (8 bf16)
        const int row   = chunk >> 2;         // 4 chunks per 32-K row
        const int c8    = chunk & 3;
        const unsigned goff  = (unsigned)(((srcRowBase + row) * DIM + ks * 32 + c8 * 8) * 2);
        const unsigned laddr = ldsBase + (unsigned)((row * TSTRIDE + c8 * 8) * 2);
        asm volatile("global_load_async_to_lds_b128 %0, %1, %2"
                     :: "v"(laddr), "v"(goff), "s"(srcBase) : "memory");
    }
}

__global__ __launch_bounds__(256) void gemm_hinge_bf16_kernel(const unsigned short* __restrict__ senBF,
                                                              const unsigned short* __restrict__ imgBF,
                                                              const float* __restrict__ diag,
                                                              float* __restrict__ partial) {
    __shared__ unsigned short ldsA[2][256 * TSTRIDE]; // image panel  256x32
    __shared__ unsigned short ldsB[2][128 * TSTRIDE]; // sentence panel 128x32
    __shared__ float diagRowS[256];
    __shared__ float diagColS[128];
    __shared__ float redS[256];

    const int tid  = threadIdx.x;
    const int lane = tid & 31;
    const int wid  = tid >> 5;
    const int waveRow = wid >> 1;  // 0..3 -> 64-row groups
    const int waveCol = wid & 1;   // 0..1 -> 64-col groups

    const int rowBase = blockIdx.y * 256; // image rows    (i)
    const int colBase = blockIdx.x * 128; // sentence rows (j)

    diagRowS[tid] = diag[rowBase + tid];
    if (tid < 128) diagColS[tid] = diag[colBase + tid];

    v8f acc[4][4] = {};

    // prologue: stage K-step 0 asynchronously
    async_tile_load<4>(imgBF, rowBase, 0, ldsA[0], tid);
    async_tile_load<2>(senBF, colBase, 0, ldsB[0], tid);

    const int KSTEPS = DIM / 32;
    for (int ks = 0; ks < KSTEPS; ++ks) {
        const int cur = ks & 1;

        // drain this wave's outstanding async copies (current buffer),
        // then barrier: all waves' copies visible, and nobody still reads
        // the buffer we are about to overwrite.
        asm volatile("s_wait_asynccnt 0x0" ::: "memory");
        __syncthreads();

        if (ks + 1 < KSTEPS) {
            async_tile_load<4>(imgBF, rowBase, ks + 1, ldsA[cur ^ 1], tid);
            async_tile_load<2>(senBF, colBase, ks + 1, ldsB[cur ^ 1], tid);
        }

        v16bf bfrag[4];
        #pragma unroll
        for (int nt = 0; nt < 4; ++nt)
            bfrag[nt] = load_frag(ldsB[cur], waveCol * 64 + nt * 16, lane);

        #pragma unroll
        for (int mt = 0; mt < 4; ++mt) {
            const v16bf afrag = load_frag(ldsA[cur], waveRow * 64 + mt * 16, lane);
            #pragma unroll
            for (int nt = 0; nt < 4; ++nt) {
                acc[mt][nt] = __builtin_amdgcn_wmma_f32_16x16x32_bf16(
                    false, afrag, false, bfrag[nt],
                    (short)0, acc[mt][nt], false, false);
            }
        }
    }
    __syncthreads();

    // hinge epilogue: C layout -> M = r + 8*lane[4], N = lane&15
    const int mHi  = (lane >> 4) << 3;
    const int nIdx = lane & 15;
    float sum = 0.0f;
    #pragma unroll
    for (int mt = 0; mt < 4; ++mt) {
        #pragma unroll
        for (int nt = 0; nt < 4; ++nt) {
            const int njBase = waveCol * 64 + nt * 16 + nIdx;
            const float dj = diagColS[njBase];
            const int j = colBase + njBase;
            #pragma unroll
            for (int r = 0; r < 8; ++r) {
                const int miBase = waveRow * 64 + mt * 16 + r + mHi;
                const int i = rowBase + miBase;
                const float s = acc[mt][nt][r];
                if (i != j) {
                    const float di = diagRowS[miBase];
                    sum += fmaxf(0.0f, MARGIN - dj + s)
                         + fmaxf(0.0f, MARGIN - di + s);
                }
            }
        }
    }

    redS[tid] = sum;
    __syncthreads();
    #pragma unroll
    for (int s2 = 128; s2 > 0; s2 >>= 1) {
        if (tid < s2) redS[tid] += redS[tid + s2];
        __syncthreads();
    }
    if (tid == 0) partial[blockIdx.y * gridDim.x + blockIdx.x] = redS[0];
}

// ---------------------------------------------------------------------------
// Kernel 2 (fallback): fp32 source, convert-on-the-fly, 128x128 tile,
//   8 waves x 32x64 (kept from round 1; only used if d_ws is too small)
// ---------------------------------------------------------------------------
__device__ __forceinline__ void load_tile_f32(const float* __restrict__ src,
                                              int srcRowBase, int ks,
                                              unsigned short* dst, int tid) {
    #pragma unroll
    for (int q = 0; q < 4; ++q) {
        const int chunk = tid + q * 256;
        const int row   = chunk >> 3;
        const int c4    = chunk & 7;
        const float4 f = *(const float4*)(src + (size_t)(srcRowBase + row) * DIM
                                          + ks * 32 + c4 * 4);
        const unsigned int p0 = (unsigned int)f2bf_rne(f.x) |
                                ((unsigned int)f2bf_rne(f.y) << 16);
        const unsigned int p1 = (unsigned int)f2bf_rne(f.z) |
                                ((unsigned int)f2bf_rne(f.w) << 16);
        unsigned int* d = (unsigned int*)(dst + row * TSTRIDE + c4 * 4);
        d[0] = p0;
        d[1] = p1;
    }
}

__global__ __launch_bounds__(256) void gemm_hinge_f32_kernel(const float* __restrict__ sen,
                                                             const float* __restrict__ img,
                                                             const float* __restrict__ diag,
                                                             float* __restrict__ partial) {
    __shared__ unsigned short ldsA[2][128 * TSTRIDE];
    __shared__ unsigned short ldsB[2][128 * TSTRIDE];
    __shared__ float diagRowS[128];
    __shared__ float diagColS[128];
    __shared__ float redS[256];

    const int tid  = threadIdx.x;
    const int lane = tid & 31;
    const int wid  = tid >> 5;
    const int waveRow = wid & 3;
    const int waveCol = wid >> 2;

    const int rowBase = blockIdx.y * 128;
    const int colBase = blockIdx.x * 128;

    if (tid < 128)       diagRowS[tid]       = diag[rowBase + tid];
    else                 diagColS[tid - 128] = diag[colBase + (tid - 128)];

    v8f acc[2][4] = {};

    load_tile_f32(img, rowBase, 0, ldsA[0], tid);
    load_tile_f32(sen, colBase, 0, ldsB[0], tid);
    __syncthreads();

    const int KSTEPS = DIM / 32;
    for (int ks = 0; ks < KSTEPS; ++ks) {
        const int cur = ks & 1;
        if (ks + 1 < KSTEPS) {
            load_tile_f32(img, rowBase, ks + 1, ldsA[cur ^ 1], tid);
            load_tile_f32(sen, colBase, ks + 1, ldsB[cur ^ 1], tid);
        }

        v16bf bfrag[4];
        #pragma unroll
        for (int nt = 0; nt < 4; ++nt)
            bfrag[nt] = load_frag(ldsB[cur], waveCol * 64 + nt * 16, lane);

        #pragma unroll
        for (int mt = 0; mt < 2; ++mt) {
            const v16bf afrag = load_frag(ldsA[cur], waveRow * 32 + mt * 16, lane);
            #pragma unroll
            for (int nt = 0; nt < 4; ++nt) {
                acc[mt][nt] = __builtin_amdgcn_wmma_f32_16x16x32_bf16(
                    false, afrag, false, bfrag[nt],
                    (short)0, acc[mt][nt], false, false);
            }
        }
        __syncthreads();
    }

    const int mHi  = (lane >> 4) << 3;
    const int nIdx = lane & 15;
    float sum = 0.0f;
    #pragma unroll
    for (int mt = 0; mt < 2; ++mt) {
        #pragma unroll
        for (int nt = 0; nt < 4; ++nt) {
            const int njBase = waveCol * 64 + nt * 16 + nIdx;
            const float dj = diagColS[njBase];
            const int j = colBase + njBase;
            #pragma unroll
            for (int r = 0; r < 8; ++r) {
                const int miBase = waveRow * 32 + mt * 16 + r + mHi;
                const int i = rowBase + miBase;
                const float s = acc[mt][nt][r];
                if (i != j) {
                    const float di = diagRowS[miBase];
                    sum += fmaxf(0.0f, MARGIN - dj + s)
                         + fmaxf(0.0f, MARGIN - di + s);
                }
            }
        }
    }

    redS[tid] = sum;
    __syncthreads();
    #pragma unroll
    for (int s2 = 128; s2 > 0; s2 >>= 1) {
        if (tid < s2) redS[tid] += redS[tid + s2];
        __syncthreads();
    }
    if (tid == 0) partial[blockIdx.y * gridDim.x + blockIdx.x] = redS[0];
}

// ---------------------------------------------------------------------------
// Kernel 3: deterministic final reduction of per-block partials
// ---------------------------------------------------------------------------
__global__ __launch_bounds__(256) void reduce_final(const float* __restrict__ partial,
                                                    float* __restrict__ out, int n) {
    __shared__ float sh[256];
    float s = 0.0f;
    for (int k = threadIdx.x; k < n; k += 256) s += partial[k];
    sh[threadIdx.x] = s;
    __syncthreads();
    #pragma unroll
    for (int s2 = 128; s2 > 0; s2 >>= 1) {
        if (threadIdx.x < s2) sh[threadIdx.x] += sh[threadIdx.x + s2];
        __syncthreads();
    }
    if (threadIdx.x == 0) out[0] = sh[0];
}

// ---------------------------------------------------------------------------
extern "C" void kernel_launch(void* const* d_in, const int* in_sizes, int n_in,
                              void* d_out, int out_size, void* d_ws, size_t ws_size,
                              hipStream_t stream) {
    (void)in_sizes; (void)n_in; (void)out_size;
    const float* sentence = (const float*)d_in[0];
    const float* image    = (const float*)d_in[1];
    float* out = (float*)d_out;

    const size_t matShorts = (size_t)N_ROWS * DIM;                 // 8.4M bf16 per matrix
    const size_t needBig   = 2 * matShorts * sizeof(unsigned short)
                           + (N_ROWS + 4096) * sizeof(float);      // ~33.6 MB

    if (ws_size >= needBig) {
        unsigned short* imgBF = (unsigned short*)d_ws;
        unsigned short* senBF = imgBF + matShorts;
        float* diag    = (float*)(senBF + matShorts);
        float* partial = diag + N_ROWS;

        diag_kernel<<<N_ROWS / 8, 256, 0, stream>>>(image, sentence, diag);
        const int cvtBlocks = (int)(matShorts / (256 * 4));        // 8192
        convert_bf16_kernel<<<cvtBlocks, 256, 0, stream>>>(image, imgBF);
        convert_bf16_kernel<<<cvtBlocks, 256, 0, stream>>>(sentence, senBF);

        const dim3 gemmGrid(N_ROWS / 128, N_ROWS / 256);           // 64 x 32
        gemm_hinge_bf16_kernel<<<gemmGrid, 256, 0, stream>>>(senBF, imgBF, diag, partial);
        reduce_final<<<1, 256, 0, stream>>>(partial, out, 64 * 32);
    } else {
        float* diag    = (float*)d_ws;
        float* partial = diag + N_ROWS;

        diag_kernel<<<N_ROWS / 8, 256, 0, stream>>>(image, sentence, diag);
        const dim3 gemmGrid(N_ROWS / 128, N_ROWS / 128);           // 64 x 64
        gemm_hinge_f32_kernel<<<gemmGrid, 256, 0, stream>>>(sentence, image, diag, partial);
        reduce_final<<<1, 256, 0, stream>>>(partial, out, 64 * 64);
    }
}
